// BucketedGoWatti_44032004719133
// MI455X (gfx1250) — matrix operand order repaired and checked by reading, not compiled, not run on hardware
//
#include <hip/hip_runtime.h>
#include <hip/hip_bf16.h>

// Problem constants (match reference)
#define BB 4
#define LL 4096
#define TT 32
#define DHH 1024
#define DGG 256
#define DPP 256
#define WINN 1024
#define STRIDEE 256
#define WW 13       // (L - WIN)/STRIDE + 1
#define KCHUNK 128  // K staging chunk for LDS-fed GEMM (double buffered: 2 x 16KB)

typedef __attribute__((ext_vector_type(16))) __bf16 v16bf;
typedef __attribute__((ext_vector_type(8)))  float  v8f;

union ABfrag { v16bf v; unsigned int u[8]; };

__device__ __forceinline__ unsigned short f2bf(float x) {
    unsigned int u = __float_as_uint(x);
    unsigned int r = (u + 0x7FFFu + ((u >> 16) & 1u)) >> 16;
    return (unsigned short)r;
}

// A fragment (16x32 bf16, row m per lane): lanes 0-15 K=0..15, lanes 16-31 K=16..31
__device__ __forceinline__ v16bf load_afrag(const unsigned short* __restrict__ A, size_t lda,
                                            int m, int kb, int hi) {
    ABfrag a;
#pragma unroll
    for (int j = 0; j < 8; ++j) {
        int k0 = (hi << 4) + (j << 1);
        a.u[j] = *(const unsigned int*)(A + (size_t)m * lda + kb + k0);
    }
    return a.v;
}

// B fragment from BT layout (element (k,n) at BT[n*ldbt+k]): lanes 0-15 K=0..15, lanes 16-31 K=16..31
__device__ __forceinline__ v16bf load_bfrag(const unsigned short* __restrict__ BT, size_t ldbt,
                                            int n, int kb, int hi) {
    ABfrag b;
#pragma unroll
    for (int j = 0; j < 8; ++j)
        b.u[j] = *(const unsigned int*)(BT + (size_t)n * ldbt + kb + (hi << 4) + (j << 1));
    return b.v;
}

// 16x64 strip per wave: A fragment reused across 4 B tiles (4x less A traffic, 4 wmma / k-step)
__device__ __forceinline__ void wmma_strip_global(const unsigned short* __restrict__ A, size_t lda,
                                                  const unsigned short* __restrict__ BT, size_t ldbt,
                                                  int K, int lane, v8f acc[4]) {
    const int m = lane & 15, hi = lane >> 4;
    for (int kb = 0; kb < K; kb += 32) {
        v16bf a = load_afrag(A, lda, m, kb, hi);
        __builtin_prefetch(A + (size_t)m * lda + kb + 128, 0, 1);
#pragma unroll
        for (int s = 0; s < 4; ++s) {
            v16bf b = load_bfrag(BT, ldbt, (s << 4) + m, kb, hi);
            acc[s] = __builtin_amdgcn_wmma_f32_16x16x32_bf16(false, a, false, b,
                                                             (short)0, acc[s], false, false);
        }
    }
}

// ---------------- converters ----------------

__global__ void k_convert_H(const float* __restrict__ H,
                            unsigned short* __restrict__ Hb,
                            unsigned short* __restrict__ HbT) {
    size_t idx = (size_t)blockIdx.x * blockDim.x + threadIdx.x;
    unsigned short v = f2bf(H[idx]);
    Hb[idx] = v;
    size_t b = idx / ((size_t)LL * DHH);
    size_t rem = idx - b * (size_t)LL * DHH;
    size_t l = rem / DHH, d = rem - l * DHH;
    HbT[b * (size_t)DHH * LL + d * (size_t)LL + l] = v;
}

__global__ void k_transpose_convert(const float* __restrict__ src,
                                    unsigned short* __restrict__ dst,
                                    int rows, int cols) {
    size_t idx = (size_t)blockIdx.x * blockDim.x + threadIdx.x;
    int r = (int)(idx / cols), c = (int)(idx - (size_t)r * cols);
    dst[(size_t)c * rows + r] = f2bf(src[idx]);
}

// ---------------- small projections: q (bf16) and qw (f32) ----------------
__global__ void k_qqw(const float* __restrict__ G,
                      const float* __restrict__ Wq_core,
                      const float* __restrict__ Wq_win,
                      unsigned short* __restrict__ qb,
                      float* __restrict__ qwf) {
    __shared__ float g[DGG];
    int bt = blockIdx.x;
    int tid = threadIdx.x;
    g[tid] = G[(size_t)bt * DGG + tid];
    __syncthreads();
    {
        float s = 0.f;
        for (int i = 0; i < DGG; ++i) s += g[i] * Wq_core[(size_t)i * DPP + tid];
        qb[(size_t)bt * DPP + tid] = f2bf(s);
    }
    for (int d = tid; d < DHH; d += blockDim.x) {
        float s = 0.f;
        for (int i = 0; i < DGG; ++i) s += g[i] * Wq_win[(size_t)i * DHH + d];
        qwf[(size_t)bt * DHH + d] = s;
    }
}

// async-stage one 64 x KCHUNK bf16 B panel chunk into LDS (all 256 threads, uniform trip count)
__device__ __forceinline__ void stage_panelB(const unsigned short* __restrict__ WkcT,
                                             unsigned short* sBbuf, int ns, int kc) {
    for (int i = threadIdx.x; i < 64 * (KCHUNK / 8); i += 256) {
        int n  = i / (KCHUNK / 8);
        int k8 = (i % (KCHUNK / 8)) * 8;
        unsigned long long src =
            (unsigned long long)(WkcT + ((size_t)(ns * 64 + n)) * DHH + kc + k8);
        unsigned ldsa = (unsigned)(size_t)&sBbuf[n * KCHUNK + k8];  // addr[31:0] = LDS offset
        asm volatile("global_load_async_to_lds_b128 %0, %1, off"
                     :: "v"(ldsa), "v"(src) : "memory");
    }
}

// ---------------- GEMM 1: K = H @ Wk_core (flagship: double-buffered async-LDS B panel) ------
// grid: (Mtiles/8 = 128, Nstrips = 4); 8 waves, one 16x64 strip each, B panel shared via LDS.
__global__ void __launch_bounds__(256) k_gemm_K(const unsigned short* __restrict__ Hb,
                                                const unsigned short* __restrict__ WkcT,
                                                unsigned short* __restrict__ Kb) {
    __shared__ unsigned short sB[2][64 * KCHUNK];  // 2 x 16 KB double buffer, BT layout
    const int lane = threadIdx.x & 31, wave = threadIdx.x >> 5;
    const int tm = blockIdx.x * 8 + wave;          // M tile index
    const int ns = blockIdx.y;                     // 64-col N strip
    const int m = lane & 15, hi = lane >> 4;
    v8f acc[4] = {{}, {}, {}, {}};
    const unsigned short* Arow = Hb + ((size_t)tm * 16 + m) * DHH;

    stage_panelB(WkcT, sB[0], ns, 0);              // prologue: chunk 0 -> buf 0

    for (int kc = 0; kc < DHH; kc += KCHUNK) {
        const int cur = (kc / KCHUNK) & 1;
        // drain this wave's outstanding async loads (== loads for buf[cur]), then block-sync
        asm volatile("s_wait_asynccnt 0x0" ::: "memory");
        __syncthreads();
        // overlap: kick off next chunk into the other buffer while we compute on this one.
        // (safe: the barrier above also proves every wave finished READING that buffer
        //  during the previous iteration's compute phase)
        if (kc + KCHUNK < DHH)
            stage_panelB(WkcT, sB[cur ^ 1], ns, kc + KCHUNK);

        for (int kb = 0; kb < KCHUNK; kb += 32) {
            ABfrag a;
#pragma unroll
            for (int j = 0; j < 8; ++j) {
                int k0 = (hi << 4) + (j << 1);
                a.u[j] = *(const unsigned int*)(Arow + kc + kb + k0);
            }
            __builtin_prefetch(Arow + kc + kb + 128, 0, 1);
#pragma unroll
            for (int s = 0; s < 4; ++s) {
                ABfrag b;
#pragma unroll
                for (int j = 0; j < 8; ++j)
                    b.u[j] = *(const unsigned int*)
                        &sB[cur][(((s << 4) + m) * KCHUNK) + kb + (hi << 4) + (j << 1)];
                acc[s] = __builtin_amdgcn_wmma_f32_16x16x32_bf16(false, a.v, false, b.v,
                                                                 (short)0, acc[s], false, false);
            }
        }
    }
    unsigned short* C = Kb + (size_t)tm * 16 * DPP + ns * 64;
#pragma unroll
    for (int s = 0; s < 4; ++s)
#pragma unroll
        for (int v = 0; v < 8; ++v)
            C[(size_t)(hi * 8 + v) * DPP + s * 16 + m] = f2bf(acc[s][v]);
}

// ---------------- GEMM 2: S = q @ K^T (per batch, strip; out f32 pre-scaled 1/16) ----------------
__global__ void k_gemm_S(const unsigned short* __restrict__ qb,
                         const unsigned short* __restrict__ Kb,
                         float* __restrict__ S) {
    const int b = blockIdx.z;
    const int lane = threadIdx.x & 31, wave = threadIdx.x >> 5;
    const int sid = blockIdx.x * 8 + wave;      // 2 (T/16) x 64 (L/64) strips
    const int ns = sid & 63;
    const int tmi = sid >> 6;
    v8f acc[4] = {{}, {}, {}, {}};
    wmma_strip_global(qb + (size_t)b * TT * DPP + (size_t)tmi * 16 * DPP, DPP,
                      Kb + (size_t)b * LL * DPP + (size_t)ns * 64 * DPP, DPP,
                      DPP, lane, acc);
    const int hi = lane >> 4, n = lane & 15;
    float* C = S + (size_t)b * TT * LL + (size_t)tmi * 16 * LL + ns * 64;
#pragma unroll
    for (int s = 0; s < 4; ++s)
#pragma unroll
        for (int v = 0; v < 8; ++v)
            C[(size_t)(hi * 8 + v) * LL + s * 16 + n] = acc[s][v] * 0.0625f;
}

// ---------------- per-window softmax -> alpha bf16 ((b,w,t), WIN) ----------------
__global__ void k_softmax_alpha(const float* __restrict__ S,
                                const unsigned char* __restrict__ mask,
                                unsigned short* __restrict__ alphab) {
    __shared__ float red[256];
    const int tid = threadIdx.x;
    const int t = blockIdx.x % TT;
    const int w = (blockIdx.x / TT) % WW;
    const int b = blockIdx.x / (TT * WW);
    const int sw = w * STRIDEE;
    const float* Srow = S + ((size_t)b * TT + t) * LL + sw;
    const unsigned char* mrow = mask + (size_t)b * LL + sw;
    float xv[WINN / 256];
    float mx = -3.402823e38f;
#pragma unroll
    for (int i = 0; i < WINN / 256; ++i) {
        int l = tid + i * 256;
        float x = mrow[l] ? Srow[l] : -1.0e9f;
        xv[i] = x;
        mx = fmaxf(mx, x);
    }
    red[tid] = mx; __syncthreads();
    for (int s = 128; s > 0; s >>= 1) { if (tid < s) red[tid] = fmaxf(red[tid], red[tid + s]); __syncthreads(); }
    mx = red[0]; __syncthreads();
    float sum = 0.f;
#pragma unroll
    for (int i = 0; i < WINN / 256; ++i) { xv[i] = __expf(xv[i] - mx); sum += xv[i]; }
    red[tid] = sum; __syncthreads();
    for (int s = 128; s > 0; s >>= 1) { if (tid < s) red[tid] += red[tid + s]; __syncthreads(); }
    float inv = 1.f / red[0];
    unsigned short* arow = alphab + (size_t)blockIdx.x * WINN;
#pragma unroll
    for (int i = 0; i < WINN / 256; ++i) arow[tid + i * 256] = f2bf(xv[i] * inv);
}

// ---------------- GEMM 3: Zw = alpha @ Hw (per (b,w), strip; out f32 + bf16) ----------------
__global__ void k_gemm_Zw(const unsigned short* __restrict__ alphab,
                          const unsigned short* __restrict__ HbT,
                          float* __restrict__ Zwf,
                          unsigned short* __restrict__ Zwb) {
    const int z = blockIdx.z;            // b*W + w
    const int b = z / WW, w = z % WW;
    const int lane = threadIdx.x & 31, wave = threadIdx.x >> 5;
    const int sid = blockIdx.x * 8 + wave;   // 2 (T/16) x 16 (DH/64) strips
    const int ns = sid & 15;
    const int tmi = sid >> 4;
    v8f acc[4] = {{}, {}, {}, {}};
    wmma_strip_global(alphab + (size_t)z * TT * WINN + (size_t)tmi * 16 * WINN, WINN,
                      HbT + (size_t)b * DHH * LL + (size_t)w * STRIDEE + (size_t)ns * 64 * LL, LL,
                      WINN, lane, acc);
    const int hi = lane >> 4, n = lane & 15;
#pragma unroll
    for (int s = 0; s < 4; ++s)
#pragma unroll
        for (int v = 0; v < 8; ++v) {
            int t = tmi * 16 + hi * 8 + v;
            int d = ns * 64 + s * 16 + n;
            size_t o = (((size_t)b * TT + t) * WW + w) * DHH + d;   // (b,t,w,d)
            Zwf[o] = acc[s][v];
            Zwb[o] = f2bf(acc[s][v]);
        }
}

// ---------------- GEMM 4: kw = Zw @ Wk_win (per batch, strip; rows = t*W+w) ----------------
__global__ void k_gemm_kw(const unsigned short* __restrict__ Zwb,
                          const unsigned short* __restrict__ WkwT,
                          float* __restrict__ kwf) {
    const int b = blockIdx.z;
    const int lane = threadIdx.x & 31, wave = threadIdx.x >> 5;
    const int sid = blockIdx.x * 8 + wave;   // 26 (416/16) x 16 (DH/64) strips
    const int ns = sid & 15;
    const int tmi = sid >> 4;
    const size_t rowsB = (size_t)TT * WW;    // 416
    v8f acc[4] = {{}, {}, {}, {}};
    wmma_strip_global(Zwb + ((size_t)b * rowsB + (size_t)tmi * 16) * DHH, DHH,
                      WkwT + (size_t)ns * 64 * DHH, DHH, DHH, lane, acc);
    const int hi = lane >> 4, n = lane & 15;
    float* C = kwf + ((size_t)b * rowsB + (size_t)tmi * 16) * DHH + ns * 64;
#pragma unroll
    for (int s = 0; s < 4; ++s)
#pragma unroll
        for (int v = 0; v < 8; ++v)
            C[(size_t)(hi * 8 + v) * DHH + s * 16 + n] = acc[s][v];
}

// ---------------- final: window softmax + combine ----------------
__global__ void k_final(const float* __restrict__ qwf,
                        const float* __restrict__ kwf,
                        const float* __restrict__ Zwf,
                        float* __restrict__ Z) {
    __shared__ float red[256];
    __shared__ float wgt[16];
    const int tid = threadIdx.x;
    const int bt = blockIdx.x;
    const int b = bt / TT, t = bt % TT;
    const float* qrow = qwf + (size_t)bt * DHH;
    for (int w = 0; w < WW; ++w) {
        float p = 0.f;
        const float* krow = kwf + ((size_t)b * TT * WW + (size_t)t * WW + w) * DHH;
        for (int d = tid; d < DHH; d += 256) p += qrow[d] * krow[d];
        red[tid] = p; __syncthreads();
        for (int s = 128; s > 0; s >>= 1) { if (tid < s) red[tid] += red[tid + s]; __syncthreads(); }
        if (tid == 0) wgt[w] = red[0] * 0.03125f;
        __syncthreads();
    }
    if (tid == 0) {
        float mx = wgt[0];
        for (int w = 1; w < WW; ++w) mx = fmaxf(mx, wgt[w]);
        float s = 0.f;
        for (int w = 0; w < WW; ++w) { wgt[w] = __expf(wgt[w] - mx); s += wgt[w]; }
        float inv = 1.f / s;
        for (int w = 0; w < WW; ++w) wgt[w] *= inv;
    }
    __syncthreads();
    for (int d = tid; d < DHH; d += 256) {
        float acc = 0.f;
        const float* zrow = Zwf + ((size_t)bt * WW) * DHH + d;
#pragma unroll
        for (int w = 0; w < WW; ++w) acc += wgt[w] * zrow[(size_t)w * DHH];
        Z[(size_t)bt * DHH + d] = acc;
    }
}

extern "C" void kernel_launch(void* const* d_in, const int* in_sizes, int n_in,
                              void* d_out, int out_size, void* d_ws, size_t ws_size,
                              hipStream_t stream) {
    const float* H       = (const float*)d_in[0];
    const float* G       = (const float*)d_in[1];
    const float* Wq_core = (const float*)d_in[2];
    const float* Wk_core = (const float*)d_in[3];
    const float* Wq_win  = (const float*)d_in[4];
    const float* Wk_win  = (const float*)d_in[5];
    const unsigned char* mask = (const unsigned char*)d_in[6];
    float* Z = (float*)d_out;

    size_t off = 0;
    char* base = (char*)d_ws;
    auto carve = [&](size_t bytes) -> void* {
        void* p = base + off;
        off = (off + bytes + 255) & ~(size_t)255;
        return p;
    };
    unsigned short* Hb    = (unsigned short*)carve((size_t)BB * LL * DHH * 2);
    unsigned short* HbT   = (unsigned short*)carve((size_t)BB * LL * DHH * 2);
    unsigned short* WkcT  = (unsigned short*)carve((size_t)DHH * DPP * 2);
    unsigned short* WkwT  = (unsigned short*)carve((size_t)DHH * DHH * 2);
    unsigned short* qb    = (unsigned short*)carve((size_t)BB * TT * DPP * 2);
    float*          qwf   = (float*)carve((size_t)BB * TT * DHH * 4);
    unsigned short* Kb    = (unsigned short*)carve((size_t)BB * LL * DPP * 2);
    float*          S     = (float*)carve((size_t)BB * TT * LL * 4);
    unsigned short* alphab= (unsigned short*)carve((size_t)BB * WW * TT * WINN * 2);
    float*          Zwf   = (float*)carve((size_t)BB * TT * WW * DHH * 4);
    unsigned short* Zwb   = (unsigned short*)carve((size_t)BB * TT * WW * DHH * 2);
    float*          kwf   = (float*)carve((size_t)BB * TT * WW * DHH * 4);

    k_convert_H<<<(BB * LL * DHH) / 256, 256, 0, stream>>>(H, Hb, HbT);
    k_transpose_convert<<<(DHH * DPP) / 256, 256, 0, stream>>>(Wk_core, WkcT, DHH, DPP);
    k_transpose_convert<<<(DHH * DHH) / 256, 256, 0, stream>>>(Wk_win, WkwT, DHH, DHH);

    k_qqw<<<BB * TT, 256, 0, stream>>>(G, Wq_core, Wq_win, qb, qwf);

    // K = H @ Wk_core : 128 M-blocks x 4 N-strips, double-buffered async-LDS B panel
    k_gemm_K<<<dim3((BB * LL / 16) / 8, DPP / 64), 256, 0, stream>>>(Hb, WkcT, Kb);

    // S = q @ K^T
    k_gemm_S<<<dim3(((TT / 16) * (LL / 64)) / 8, 1, BB), 256, 0, stream>>>(qb, Kb, S);

    k_softmax_alpha<<<BB * WW * TT, 256, 0, stream>>>(S, mask, alphab);

    // Zw = alpha @ Hw
    k_gemm_Zw<<<dim3(((TT / 16) * (DHH / 64)) / 8, 1, BB * WW), 256, 0, stream>>>(alphab, HbT, Zwf, Zwb);

    // kw = Zw @ Wk_win
    k_gemm_kw<<<dim3(((TT * WW / 16) * (DHH / 64)) / 8, 1, BB), 256, 0, stream>>>(Zwb, WkwT, kwf);

    k_final<<<BB * TT, 256, 0, stream>>>(qwf, kwf, Zwf, Z);
}
